// DeformableDecoder_79869211836817
// MI455X (gfx1250) — matrix operand order repaired and checked by reading, not compiled, not run on hardware
//
#include <hip/hip_runtime.h>

#define BB 4
#define QMAX 1000
#define DMODEL 256
#define NHEAD 8
#define HDIM 32
#define FFDIM 1024
#define NLVL 4
#define NPT 4
#define NLAYER 6

typedef __bf16 bf16_t;
typedef __bf16 v2bf  __attribute__((ext_vector_type(2)));
typedef __bf16 v16bf __attribute__((ext_vector_type(16)));
typedef float  v8f   __attribute__((ext_vector_type(8)));

static __device__ __forceinline__ int imin(int a, int b) { return a < b ? a : b; }

// pack two f32 -> packed bf16 pair (lowers to v_cvt_pk_bf16_f32)
static __device__ __forceinline__ unsigned int pkbf(float x, float y)
{
    v2bf p;
    p[0] = (bf16_t)x;
    p[1] = (bf16_t)y;
    return __builtin_bit_cast(unsigned int, p);
}

// ---------------------------------------------------------------------------
// Generic GEMM: C[M,N] = act(A[M,K] @ W[N,K]^T + bias) (+ residual)
// act: 0=none 1=relu 2=tanh.  K % 32 == 0, N % 128 == 0.
// 256 threads, tile 64x128. 8 waves: each computes a 32x32 sub-tile as
// 2x2 WMMA 16x16x32 (2 A-frags x 2 B-frags -> 4 wmma per K step).
// ---------------------------------------------------------------------------
__global__ __launch_bounds__(256) void gemm_kernel(
    const float* __restrict__ A, int M, int K,
    const float* __restrict__ W, const float* __restrict__ bias,
    float* __restrict__ C, int N,
    const float* __restrict__ residual, int act)
{
    __shared__ unsigned int Asu[64][20];   // 64 rows x 32 bf16 (16 uints) + pad
    __shared__ unsigned int Bsu[128][20];  // 128 rows x 32 bf16 + pad
    int tid = threadIdx.x, wave = tid >> 5, lane = tid & 31;
    int wm = wave >> 2, wn = wave & 3;         // 2 x 4 wave grid
    int half = lane >> 4, l16 = lane & 15;
    int m0 = blockIdx.y * 64, n0 = blockIdx.x * 128;

    int ra = tid >> 3;            // 0..31 (base row this thread stages)
    int cc = (tid & 7) * 4;       // k-column within tile (float4 granularity)

    v8f a00 = {}, a01 = {}, a10 = {}, a11 = {};

    for (int k0 = 0; k0 < K; k0 += 32) {
        // ---- stage A (64 rows) : 2 float4 per thread, branch-free ----
        int gm0 = m0 + ra, gm1 = m0 + 32 + ra;
        float4 av0 = *(const float4*)(A + (size_t)imin(gm0, M - 1) * K + k0 + cc);
        float4 av1 = *(const float4*)(A + (size_t)imin(gm1, M - 1) * K + k0 + cc);
        // ---- stage B (128 rows) : 4 float4 per thread ----
        int gn0 = n0 + ra, gn1 = gn0 + 32, gn2 = gn0 + 64, gn3 = gn0 + 96;
        float4 bv0 = *(const float4*)(W + (size_t)imin(gn0, N - 1) * K + k0 + cc);
        float4 bv1 = *(const float4*)(W + (size_t)imin(gn1, N - 1) * K + k0 + cc);
        float4 bv2 = *(const float4*)(W + (size_t)imin(gn2, N - 1) * K + k0 + cc);
        float4 bv3 = *(const float4*)(W + (size_t)imin(gn3, N - 1) * K + k0 + cc);

        float ma0 = (gm0 < M) ? 1.f : 0.f, ma1 = (gm1 < M) ? 1.f : 0.f;
        float mb0 = (gn0 < N) ? 1.f : 0.f, mb1 = (gn1 < N) ? 1.f : 0.f;
        float mb2 = (gn2 < N) ? 1.f : 0.f, mb3 = (gn3 < N) ? 1.f : 0.f;
        int cu = cc >> 1;
        Asu[ra][cu]          = pkbf(av0.x * ma0, av0.y * ma0);
        Asu[ra][cu + 1]      = pkbf(av0.z * ma0, av0.w * ma0);
        Asu[ra + 32][cu]     = pkbf(av1.x * ma1, av1.y * ma1);
        Asu[ra + 32][cu + 1] = pkbf(av1.z * ma1, av1.w * ma1);
        Bsu[ra][cu]          = pkbf(bv0.x * mb0, bv0.y * mb0);
        Bsu[ra][cu + 1]      = pkbf(bv0.z * mb0, bv0.w * mb0);
        Bsu[ra + 32][cu]     = pkbf(bv1.x * mb1, bv1.y * mb1);
        Bsu[ra + 32][cu + 1] = pkbf(bv1.z * mb1, bv1.w * mb1);
        Bsu[ra + 64][cu]     = pkbf(bv2.x * mb2, bv2.y * mb2);
        Bsu[ra + 64][cu + 1] = pkbf(bv2.z * mb2, bv2.w * mb2);
        Bsu[ra + 96][cu]     = pkbf(bv3.x * mb3, bv3.y * mb3);
        Bsu[ra + 96][cu + 1] = pkbf(bv3.z * mb3, bv3.w * mb3);

        if (k0 + 32 < K) {
            __builtin_prefetch(A + (size_t)imin(gm0, M - 1) * K + k0 + 32 + cc, 0, 1);
            __builtin_prefetch(W + (size_t)imin(gn0, N - 1) * K + k0 + 32 + cc, 0, 1);
        }
        __syncthreads();

        // ---- fragments + 4 WMMA ----
        const bf16_t* ar0 = (const bf16_t*)&Asu[32 * wm + l16][0];
        const bf16_t* ar1 = (const bf16_t*)&Asu[32 * wm + 16 + l16][0];
        const bf16_t* br0 = (const bf16_t*)&Bsu[32 * wn + l16][0];
        const bf16_t* br1 = (const bf16_t*)&Bsu[32 * wn + 16 + l16][0];
        v16bf af0, af1, bf0, bf1;
        #pragma unroll
        for (int e = 0; e < 16; e++) {
            int ka = e + 8 * half + ((e >= 8) ? 8 : 0);
            int kb = e + 8 * half + ((e >= 8) ? 8 : 0);
            af0[e] = ar0[ka]; af1[e] = ar1[ka];
            bf0[e] = br0[kb]; bf1[e] = br1[kb];
        }
        a00 = __builtin_amdgcn_wmma_f32_16x16x32_bf16(false, af0, false, bf0, (short)0, a00, false, false);
        a01 = __builtin_amdgcn_wmma_f32_16x16x32_bf16(false, af0, false, bf1, (short)0, a01, false, false);
        a10 = __builtin_amdgcn_wmma_f32_16x16x32_bf16(false, af1, false, bf0, (short)0, a10, false, false);
        a11 = __builtin_amdgcn_wmma_f32_16x16x32_bf16(false, af1, false, bf1, (short)0, a11, false, false);
        __syncthreads();
    }

    #pragma unroll
    for (int tm = 0; tm < 2; tm++) {
        #pragma unroll
        for (int tn = 0; tn < 2; tn++) {
            v8f acc = (tm == 0) ? (tn == 0 ? a00 : a01) : (tn == 0 ? a10 : a11);
            int col = n0 + 32 * wn + 16 * tn + l16;
            #pragma unroll
            for (int r = 0; r < 8; r++) {
                int row = m0 + 32 * wm + 16 * tm + r + 8 * half;
                if (row < M && col < N) {
                    float v = acc[r] + (bias ? bias[col] : 0.0f);
                    if (act == 1) v = fmaxf(v, 0.0f);
                    else if (act == 2) v = tanhf(v);
                    if (residual) v += residual[(size_t)row * N + col];
                    C[(size_t)row * N + col] = v;
                }
            }
        }
    }
}

// ---------------------------------------------------------------------------
// LayerNorm over D=256; writes ln and ln + q_embed. one wave per row,
// contiguous 8 floats per lane (float4 loads/stores).
// ---------------------------------------------------------------------------
__global__ __launch_bounds__(32) void ln_kernel(
    const float* __restrict__ x, const float* __restrict__ w,
    const float* __restrict__ b, const float* __restrict__ qe,
    float* __restrict__ ln, float* __restrict__ lnpe)
{
    int row = blockIdx.x, lane = threadIdx.x;
    int base = lane * 8;
    const float* xr = x + (size_t)row * DMODEL + base;
    float v[8];
    *(float4*)&v[0] = *(const float4*)xr;
    *(float4*)&v[4] = *(const float4*)(xr + 4);
    float s = 0.f;
    #pragma unroll
    for (int j = 0; j < 8; j++) s += v[j];
    #pragma unroll
    for (int m = 16; m > 0; m >>= 1) s += __shfl_xor(s, m, 32);
    float mu = s * (1.0f / DMODEL), ss = 0.f;
    #pragma unroll
    for (int j = 0; j < 8; j++) { float d = v[j] - mu; ss += d * d; }
    #pragma unroll
    for (int m = 16; m > 0; m >>= 1) ss += __shfl_xor(ss, m, 32);
    float rstd = rsqrtf(ss * (1.0f / DMODEL) + 1e-5f);
    float wv[8], bv[8], qv[8], y[8], yq[8];
    *(float4*)&wv[0] = *(const float4*)(w + base);
    *(float4*)&wv[4] = *(const float4*)(w + base + 4);
    *(float4*)&bv[0] = *(const float4*)(b + base);
    *(float4*)&bv[4] = *(const float4*)(b + base + 4);
    const float* qr = qe + (size_t)row * DMODEL + base;
    *(float4*)&qv[0] = *(const float4*)qr;
    *(float4*)&qv[4] = *(const float4*)(qr + 4);
    #pragma unroll
    for (int j = 0; j < 8; j++) {
        y[j] = (v[j] - mu) * rstd * wv[j] + bv[j];
        yq[j] = y[j] + qv[j];
    }
    float* lr = ln + (size_t)row * DMODEL + base;
    float* pr = lnpe + (size_t)row * DMODEL + base;
    *(float4*)lr = *(float4*)&y[0];       *(float4*)(lr + 4) = *(float4*)&y[4];
    *(float4*)pr = *(float4*)&yq[0];      *(float4*)(pr + 4) = *(float4*)&yq[4];
}

// ---------------------------------------------------------------------------
// Flash-style MHA: one wave per (b, head, 16-row q tile). HD=32 == WMMA K.
// qk: [B*Q, 512] (q cols 0..255, k cols 256..511), v: [B*Q, 256].
// ---------------------------------------------------------------------------
__global__ __launch_bounds__(32) void attn_kernel(
    const float* __restrict__ qk, const float* __restrict__ vbuf,
    float* __restrict__ out, int Q)
{
    int qtc = (Q + 15) / 16;
    int bid = blockIdx.x;
    int qt = bid % qtc; bid /= qtc;
    int h = bid % NHEAD, b = bid / NHEAD;
    int lane = threadIdx.x;
    int half = lane >> 4, l16 = lane & 15;
    const float scale = 0.17677669529663687f; // 1/sqrt(32)

    __shared__ unsigned int Vl[32][18];  // 32 kv rows x 32 bf16 + pad
    __shared__ bf16_t Pl[16][34];

    // Q fragment: per-lane contiguous segments (K = 16*half+e grouping)
    int qrow = imin(qt * 16 + l16, Q - 1);
    const float* qp = qk + (size_t)(b * Q + qrow) * 512 + h * 32 + 16 * half;
    float qa[16];
    *(float4*)&qa[0]  = *(const float4*)qp;
    *(float4*)&qa[4]  = *(const float4*)(qp + 4);
    *(float4*)&qa[8]  = *(const float4*)(qp + 8);
    *(float4*)&qa[12] = *(const float4*)(qp + 12);
    v16bf qf;
    #pragma unroll
    for (int e = 0; e < 16; e++) qf[e] = (bf16_t)(qa[e] * scale);

    float rm[8], rl[8];
    v8f o0 = {}, o1 = {};
    #pragma unroll
    for (int r = 0; r < 8; r++) { rm[r] = -1e30f; rl[r] = 0.0f; }

    int nkv = (Q + 31) / 32;
    for (int kt = 0; kt < nkv; kt++) {
        // K fragments: per-lane 16 contiguous floats
        int kv0 = kt * 32 + l16, kv1 = kv0 + 16;
        const float* kp0 = qk + (size_t)(b * Q + imin(kv0, Q - 1)) * 512 + 256 + h * 32 + 16 * half;
        const float* kp1 = qk + (size_t)(b * Q + imin(kv1, Q - 1)) * 512 + 256 + h * 32 + 16 * half;
        float ka0[16], ka1[16];
        *(float4*)&ka0[0]  = *(const float4*)kp0;       *(float4*)&ka0[4]  = *(const float4*)(kp0 + 4);
        *(float4*)&ka0[8]  = *(const float4*)(kp0 + 8); *(float4*)&ka0[12] = *(const float4*)(kp0 + 12);
        *(float4*)&ka1[0]  = *(const float4*)kp1;       *(float4*)&ka1[4]  = *(const float4*)(kp1 + 4);
        *(float4*)&ka1[8]  = *(const float4*)(kp1 + 8); *(float4*)&ka1[12] = *(const float4*)(kp1 + 12);
        v16bf kf0, kf1;
        #pragma unroll
        for (int e = 0; e < 16; e++) { kf0[e] = (bf16_t)ka0[e]; kf1[e] = (bf16_t)ka1[e]; }

        // stage V tile in LDS (vectorized, masked)
        int kvv = kt * 32 + lane;
        const float* vp = vbuf + (size_t)(b * Q + imin(kvv, Q - 1)) * 256 + h * 32;
        float mv = (kvv < Q) ? 1.f : 0.f;
        #pragma unroll
        for (int j = 0; j < 8; j++) {
            float4 f = *(const float4*)(vp + 4 * j);
            Vl[lane][2 * j]     = pkbf(f.x * mv, f.y * mv);
            Vl[lane][2 * j + 1] = pkbf(f.z * mv, f.w * mv);
        }

        v8f s0 = {}, s1 = {};
        s0 = __builtin_amdgcn_wmma_f32_16x16x32_bf16(false, qf, false, kf0, (short)0, s0, false, false);
        s1 = __builtin_amdgcn_wmma_f32_16x16x32_bf16(false, qf, false, kf1, (short)0, s1, false, false);

        int kvA = kt * 32 + l16, kvB = kvA + 16;
        #pragma unroll
        for (int r = 0; r < 8; r++) {
            float a = (kvA < Q) ? s0[r] : -1e30f;
            float c = (kvB < Q) ? s1[r] : -1e30f;
            float mx = fmaxf(a, c);
            #pragma unroll
            for (int msk = 1; msk < 16; msk <<= 1) mx = fmaxf(mx, __shfl_xor(mx, msk, 32));
            float nm = fmaxf(rm[r], mx);
            float sc = __expf(rm[r] - nm);
            float e0 = __expf(a - nm), e1 = __expf(c - nm);
            float rs = e0 + e1;
            #pragma unroll
            for (int msk = 1; msk < 16; msk <<= 1) rs += __shfl_xor(rs, msk, 32);
            rl[r] = rl[r] * sc + rs;
            o0[r] *= sc; o1[r] *= sc;
            rm[r] = nm;
            Pl[r + 8 * half][l16]      = (bf16_t)e0;
            Pl[r + 8 * half][16 + l16] = (bf16_t)e1;
        }
        __syncthreads();

        v16bf pf, vf0, vf1;
        #pragma unroll
        for (int e = 0; e < 16; e++) {
            int kk = e + 16 * half;
            pf[e] = Pl[l16][kk];
            const bf16_t* vrow = (const bf16_t*)&Vl[e + 16 * half][0];
            vf0[e] = vrow[l16];
            vf1[e] = vrow[16 + l16];
        }
        o0 = __builtin_amdgcn_wmma_f32_16x16x32_bf16(false, pf, false, vf0, (short)0, o0, false, false);
        o1 = __builtin_amdgcn_wmma_f32_16x16x32_bf16(false, pf, false, vf1, (short)0, o1, false, false);
        __syncthreads();
    }
    #pragma unroll
    for (int r = 0; r < 8; r++) {
        int qrw = qt * 16 + r + 8 * half;
        if (qrw < Q) {
            float inv = 1.0f / rl[r];
            out[(size_t)(b * Q + qrw) * 256 + h * 32 + l16]      = o0[r] * inv;
            out[(size_t)(b * Q + qrw) * 256 + h * 32 + 16 + l16] = o1[r] * inv;
        }
    }
}

// softmax over the 16 (level,point) logits per (b,q,head); writes attn output.
__global__ void softmax16_kernel(const float* __restrict__ logits,
                                 float* __restrict__ attnout, int total)
{
    int i = blockIdx.x * blockDim.x + threadIdx.x;
    if (i >= total) return;
    const float* p = logits + (size_t)i * 16;
    float e[16], mx = -1e30f;
    *(float4*)&e[0]  = *(const float4*)p;
    *(float4*)&e[4]  = *(const float4*)(p + 4);
    *(float4*)&e[8]  = *(const float4*)(p + 8);
    *(float4*)&e[12] = *(const float4*)(p + 12);
    #pragma unroll
    for (int j = 0; j < 16; j++) mx = fmaxf(mx, e[j]);
    float s = 0.f;
    #pragma unroll
    for (int j = 0; j < 16; j++) { e[j] = __expf(e[j] - mx); s += e[j]; }
    float inv = 1.0f / s;
    #pragma unroll
    for (int j = 0; j < 16; j++) e[j] *= inv;
    float* o = attnout + (size_t)i * 16;
    *(float4*)o        = *(float4*)&e[0];
    *(float4*)(o + 4)  = *(float4*)&e[4];
    *(float4*)(o + 8)  = *(float4*)&e[8];
    *(float4*)(o + 12) = *(float4*)&e[12];
}

// deformable bilinear sampling: one wave = 32 channels of one (b,q,head).
__global__ __launch_bounds__(32) void sample_kernel(
    const float* __restrict__ ref, const float* __restrict__ off,
    const float* __restrict__ attn,
    const float* __restrict__ f0, const float* __restrict__ f1,
    const float* __restrict__ f2, const float* __restrict__ f3,
    float* __restrict__ agg, int Q)
{
    int bid = blockIdx.x;
    int h = bid % NHEAD;
    int q = (bid / NHEAD) % Q;
    int b = bid / (NHEAD * Q);
    int c = threadIdx.x;
    size_t rq = (size_t)(b * Q + q);
    float rx = ref[rq * 2 + 0] * 2.f - 1.f;
    float ry = ref[rq * 2 + 1] * 2.f - 1.f;
    const float* feats[4] = { f0, f1, f2, f3 };
    const int HS[4] = { 128, 64, 32, 16 };
    float accv = 0.f;
    for (int lvl = 0; lvl < NLVL; lvl++) {
        int H = HS[lvl], W = HS[lvl];
        const float* fch = feats[lvl] + (size_t)(b * DMODEL + h * HDIM + c) * H * W;
        #pragma unroll
        for (int p = 0; p < NPT; p++) {
            size_t oidx = rq * 256 + (size_t)(((h * NLVL + lvl) * NPT + p) * 2);
            float gx = rx + off[oidx], gy = ry + off[oidx + 1];
            float wgt = attn[(rq * NHEAD + h) * 16 + lvl * NPT + p];
            float fx = (gx + 1.f) * 0.5f * (W - 1);
            float fy = (gy + 1.f) * 0.5f * (H - 1);
            float x0f = floorf(fx), y0f = floorf(fy);
            int x0 = (int)x0f, y0 = (int)y0f;
            float wx1 = fx - x0f, wy1 = fy - y0f;
            float s = 0.f;
            if (x0 >= 0 && x0 <= W - 1 && y0 >= 0 && y0 <= H - 1)         s += fch[y0 * W + x0] * (1 - wx1) * (1 - wy1);
            if (x0 + 1 >= 0 && x0 + 1 <= W - 1 && y0 >= 0 && y0 <= H - 1) s += fch[y0 * W + x0 + 1] * wx1 * (1 - wy1);
            if (x0 >= 0 && x0 <= W - 1 && y0 + 1 >= 0 && y0 + 1 <= H - 1) s += fch[(y0 + 1) * W + x0] * (1 - wx1) * wy1;
            if (x0 + 1 >= 0 && x0 + 1 <= W - 1 && y0 + 1 >= 0 && y0 + 1 <= H - 1) s += fch[(y0 + 1) * W + x0 + 1] * wx1 * wy1;
            accv += wgt * s;
        }
    }
    agg[rq * 256 + h * HDIM + c] = accv;
}

// initial reference points: sigmoid of 2-wide projection of hidden h.
__global__ __launch_bounds__(32) void refinit_kernel(
    const float* __restrict__ h, const float* __restrict__ w2,
    const float* __restrict__ b2, float* __restrict__ ref)
{
    int row = blockIdx.x, lane = threadIdx.x;
    int base = lane * 8;
    const float* xr = h + (size_t)row * DMODEL + base;
    float x[8];
    *(float4*)&x[0] = *(const float4*)xr;
    *(float4*)&x[4] = *(const float4*)(xr + 4);
    float d0 = 0, d1 = 0;
    #pragma unroll
    for (int j = 0; j < 8; j++) { d0 += x[j] * w2[base + j]; d1 += x[j] * w2[DMODEL + base + j]; }
    #pragma unroll
    for (int m = 16; m > 0; m >>= 1) { d0 += __shfl_xor(d0, m, 32); d1 += __shfl_xor(d1, m, 32); }
    if (lane == 0) {
        ref[row * 2 + 0] = 1.f / (1.f + __expf(-(d0 + b2[0])));
        ref[row * 2 + 1] = 1.f / (1.f + __expf(-(d1 + b2[1])));
    }
}

// refine ref points (tanh*0.5, clip) and optionally compute selection scores.
__global__ __launch_bounds__(32) void rowops_kernel(
    const float* __restrict__ out, const float* __restrict__ rw,
    const float* __restrict__ rb, const float* __restrict__ sw,
    const float* __restrict__ sb, float* __restrict__ ref,
    float* __restrict__ scores, int do_score)
{
    int row = blockIdx.x, lane = threadIdx.x;
    int base = lane * 8;
    const float* xr = out + (size_t)row * DMODEL + base;
    float x[8];
    *(float4*)&x[0] = *(const float4*)xr;
    *(float4*)&x[4] = *(const float4*)(xr + 4);
    float d0 = 0, d1 = 0, ds = 0;
    #pragma unroll
    for (int j = 0; j < 8; j++) {
        d0 += x[j] * rw[base + j];
        d1 += x[j] * rw[DMODEL + base + j];
        ds += x[j] * sw[base + j];
    }
    #pragma unroll
    for (int m = 16; m > 0; m >>= 1) {
        d0 += __shfl_xor(d0, m, 32); d1 += __shfl_xor(d1, m, 32); ds += __shfl_xor(ds, m, 32);
    }
    if (lane == 0) {
        float r0 = ref[row * 2 + 0] + tanhf(d0 + rb[0]) * 0.5f;
        float r1 = ref[row * 2 + 1] + tanhf(d1 + rb[1]) * 0.5f;
        ref[row * 2 + 0] = fminf(fmaxf(r0, 0.f), 1.f);
        ref[row * 2 + 1] = fminf(fmaxf(r1, 0.f), 1.f);
        if (do_score) scores[row] = 1.f / (1.f + __expf(-(ds + sb[0])));
    }
}

// iterative arg-max top-k per batch (matches lax.top_k tie-breaking: lower idx).
__global__ void topk_kernel(const float* __restrict__ scores, int* __restrict__ idx,
                            int Q, int keep)
{
    __shared__ float sv[1024];
    __shared__ float rv[256];
    __shared__ int ri[256];
    int b = blockIdx.x, t = threadIdx.x;
    for (int j = t; j < Q; j += 256) sv[j] = scores[b * Q + j];
    __syncthreads();
    for (int it = 0; it < keep; it++) {
        float best = -1e30f; int bi = 0x7fffffff;
        for (int j = t; j < Q; j += 256) {
            float v = sv[j];
            if (v > best || (v == best && j < bi)) { best = v; bi = j; }
        }
        rv[t] = best; ri[t] = bi;
        __syncthreads();
        for (int s = 128; s > 0; s >>= 1) {
            if (t < s) {
                float ov = rv[t + s]; int oi = ri[t + s];
                if (ov > rv[t] || (ov == rv[t] && oi < ri[t])) { rv[t] = ov; ri[t] = oi; }
            }
            __syncthreads();
        }
        if (t == 0) { idx[b * keep + it] = ri[0]; sv[ri[0]] = -1e30f; }
        __syncthreads();
    }
}

// gather kept queries: 64 threads, float4 row copies.
__global__ __launch_bounds__(64) void gather_kernel(
    const float* __restrict__ so, const float* __restrict__ sq,
    const float* __restrict__ sr, const int* __restrict__ idx,
    float* __restrict__ dof, float* __restrict__ dq,
    float* __restrict__ dr, int Qold, int keep)
{
    int j = blockIdx.x;
    int b = j / keep, jj = j % keep;
    int s = idx[j];
    int t = threadIdx.x;
    const float4* s4o = (const float4*)(so + (size_t)(b * Qold + s) * DMODEL);
    const float4* s4q = (const float4*)(sq + (size_t)(b * Qold + s) * DMODEL);
    float4* d4o = (float4*)(dof + (size_t)(b * keep + jj) * DMODEL);
    float4* d4q = (float4*)(dq + (size_t)(b * keep + jj) * DMODEL);
    d4o[t] = s4o[t];
    d4q[t] = s4q[t];
    if (t < 2) dr[(b * keep + jj) * 2 + t] = sr[(b * Qold + s) * 2 + t];
}

// ---------------------------------------------------------------------------
extern "C" void kernel_launch(void* const* d_in, const int* in_sizes, int n_in,
                              void* d_out_v, int out_size, void* d_ws, size_t ws_size,
                              hipStream_t stream)
{
    (void)in_sizes; (void)n_in; (void)out_size; (void)ws_size;
    const float* q_obj   = (const float*)d_in[0];
    const float* q_emb   = (const float*)d_in[1];
    const float* feat[4] = { (const float*)d_in[2], (const float*)d_in[3],
                             (const float*)d_in[4], (const float*)d_in[5] };
    const float* sa_in_w = (const float*)d_in[6];   const float* sa_in_b = (const float*)d_in[7];
    const float* sa_out_w= (const float*)d_in[8];   const float* sa_out_b= (const float*)d_in[9];
    const float* off_w   = (const float*)d_in[10];  const float* off_b   = (const float*)d_in[11];
    const float* aw_w    = (const float*)d_in[12];  const float* aw_b    = (const float*)d_in[13];
    const float* ca_out_w= (const float*)d_in[14];  const float* ca_out_b= (const float*)d_in[15];
    const float* ff_w1   = (const float*)d_in[16];  const float* ff_b1   = (const float*)d_in[17];
    const float* ff_w2   = (const float*)d_in[18];  const float* ff_b2   = (const float*)d_in[19];
    const float* nsa_w   = (const float*)d_in[20];  const float* nsa_b   = (const float*)d_in[21];
    const float* nca_w   = (const float*)d_in[22];  const float* nca_b   = (const float*)d_in[23];
    const float* nff_w   = (const float*)d_in[24];  const float* nff_b   = (const float*)d_in[25];
    const float* ref_w1  = (const float*)d_in[26];  const float* ref_b1  = (const float*)d_in[27];
    const float* ref_w2  = (const float*)d_in[28];  const float* ref_b2  = (const float*)d_in[29];
    const float* score_w = (const float*)d_in[30];  const float* score_b = (const float*)d_in[31];
    const float* ref_rw  = (const float*)d_in[32];  const float* ref_rb  = (const float*)d_in[33];

    float* out_f = (float*)d_out_v;
    float* ws = (float*)d_ws;
    const size_t SZ = (size_t)BB * QMAX * DMODEL;
    size_t pos = 0;
    auto carve = [&](size_t n) { float* p = ws + pos; pos += n; return p; };
    float* out_a  = carve(SZ); float* out_b  = carve(SZ);
    float* qe_a   = carve(SZ); float* qe_b   = carve(SZ);
    float* lnbuf  = carve(SZ); float* lnpe   = carve(SZ);
    float* vbuf   = carve(SZ); float* attn_o = carve(SZ);
    float* ffh    = carve((size_t)BB * QMAX * FFDIM);
    float* qkbuf  = ffh;                              // [BQ,512] lives inside ffh
    float* awbuf  = ffh + (size_t)BB * QMAX * 512;    // [BQ,128] lives inside ffh
    float* ref_a  = carve((size_t)BB * QMAX * 2);
    float* ref_bb = carve((size_t)BB * QMAX * 2);
    float* scores = carve((size_t)BB * QMAX);
    int*   idxbuf = (int*)carve((size_t)BB * QMAX);

    int Qs[NLAYER]; Qs[0] = QMAX;
    for (int i = 1; i < NLAYER; i++) { int k = Qs[i - 1] / 2; Qs[i] = k < 16 ? 16 : k; }
    size_t out_off[NLAYER], attn_off[NLAYER], acc = 0;
    for (int i = 0; i < NLAYER; i++) { out_off[i] = acc; acc += (size_t)BB * Qs[i] * DMODEL; }
    for (int i = 0; i < NLAYER; i++) { attn_off[i] = acc; acc += (size_t)BB * Qs[i] * NHEAD * NLVL * NPT; }

    hipMemcpyAsync(out_a, q_obj, SZ * sizeof(float), hipMemcpyDeviceToDevice, stream);
    hipMemcpyAsync(qe_a, q_emb, SZ * sizeof(float), hipMemcpyDeviceToDevice, stream);

    { // initial reference points: sigmoid(relu(qe @ w1^T + b1) @ w2^T + b2)
        int M = BB * QMAX;
        dim3 g((DMODEL + 127) / 128, (M + 63) / 64);
        gemm_kernel<<<g, 256, 0, stream>>>(q_emb, M, DMODEL, ref_w1, ref_b1, ffh, DMODEL, nullptr, 1);
        refinit_kernel<<<M, 32, 0, stream>>>(ffh, ref_w2, ref_b2, ref_a);
    }

    float *cur_out = out_a, *alt_out = out_b;
    float *cur_qe = qe_a,  *alt_qe = qe_b;
    float *cur_ref = ref_a, *alt_ref = ref_bb;

    for (int i = 0; i < NLAYER; i++) {
        int Q = Qs[i], M = BB * Q;
        int gy = (M + 63) / 64;
        dim3 g256(2, gy), g512(4, gy), g128(1, gy), g1024(8, gy);
        // ---- self attention ----
        ln_kernel<<<M, 32, 0, stream>>>(cur_out, nsa_w + i * DMODEL, nsa_b + i * DMODEL, cur_qe, lnbuf, lnpe);
        gemm_kernel<<<g512, 256, 0, stream>>>(lnpe, M, DMODEL, sa_in_w + (size_t)i * 768 * DMODEL,
                                              sa_in_b + i * 768, qkbuf, 512, nullptr, 0);
        gemm_kernel<<<g256, 256, 0, stream>>>(lnbuf, M, DMODEL, sa_in_w + (size_t)i * 768 * DMODEL + 512 * DMODEL,
                                              sa_in_b + i * 768 + 512, vbuf, 256, nullptr, 0);
        { int qt = (Q + 15) / 16;
          attn_kernel<<<BB * NHEAD * qt, 32, 0, stream>>>(qkbuf, vbuf, attn_o, Q); }
        gemm_kernel<<<g256, 256, 0, stream>>>(attn_o, M, DMODEL, sa_out_w + (size_t)i * DMODEL * DMODEL,
                                              sa_out_b + i * DMODEL, cur_out, 256, cur_out, 0);
        // ---- deformable cross attention ----
        ln_kernel<<<M, 32, 0, stream>>>(cur_out, nca_w + i * DMODEL, nca_b + i * DMODEL, cur_qe, lnbuf, lnpe);
        gemm_kernel<<<g256, 256, 0, stream>>>(lnpe, M, DMODEL, off_w + (size_t)i * DMODEL * DMODEL,
                                              off_b + i * DMODEL, lnbuf, 256, nullptr, 2); // tanh offsets
        gemm_kernel<<<g128, 256, 0, stream>>>(lnpe, M, DMODEL, aw_w + (size_t)i * 128 * DMODEL,
                                              aw_b + i * 128, awbuf, 128, nullptr, 0);
        { int tot = M * NHEAD;
          softmax16_kernel<<<(tot + 255) / 256, 256, 0, stream>>>(awbuf, out_f + attn_off[i], tot); }
        sample_kernel<<<M * NHEAD, 32, 0, stream>>>(cur_ref, lnbuf, out_f + attn_off[i],
                                                    feat[0], feat[1], feat[2], feat[3], attn_o, Q);
        gemm_kernel<<<g256, 256, 0, stream>>>(attn_o, M, DMODEL, ca_out_w + (size_t)i * DMODEL * DMODEL,
                                              ca_out_b + i * DMODEL, cur_out, 256, cur_out, 0);
        // ---- FFN ----
        ln_kernel<<<M, 32, 0, stream>>>(cur_out, nff_w + i * DMODEL, nff_b + i * DMODEL, cur_qe, lnbuf, lnpe);
        gemm_kernel<<<g1024, 256, 0, stream>>>(lnbuf, M, DMODEL, ff_w1 + (size_t)i * FFDIM * DMODEL,
                                               ff_b1 + i * FFDIM, ffh, FFDIM, nullptr, 1);
        gemm_kernel<<<g256, 256, 0, stream>>>(ffh, M, FFDIM, ff_w2 + (size_t)i * DMODEL * FFDIM,
                                              ff_b2 + i * DMODEL, cur_out, 256, cur_out, 0);
        // layer output
        hipMemcpyAsync(out_f + out_off[i], cur_out, (size_t)M * DMODEL * sizeof(float),
                       hipMemcpyDeviceToDevice, stream);
        // refine reference points (+ scores for pruning)
        rowops_kernel<<<M, 32, 0, stream>>>(cur_out, ref_rw, ref_rb, score_w, score_b,
                                            cur_ref, scores, (i < NLAYER - 1) ? 1 : 0);
        if (i < NLAYER - 1) {
            int keep = Qs[i + 1];
            topk_kernel<<<BB, 256, 0, stream>>>(scores, idxbuf, Q, keep);
            gather_kernel<<<BB * keep, 64, 0, stream>>>(cur_out, cur_qe, cur_ref, idxbuf,
                                                        alt_out, alt_qe, alt_ref, Q, keep);
            float* t;
            t = cur_out; cur_out = alt_out; alt_out = t;
            t = cur_qe;  cur_qe = alt_qe;   alt_qe = t;
            t = cur_ref; cur_ref = alt_ref; alt_ref = t;
        }
    }
}
